// TMIL_62586263437580
// MI455X (gfx1250) — matrix operand order repaired and checked by reading, not compiled, not run on hardware
//
#include <hip/hip_runtime.h>
#include <hip/hip_bf16.h>
#include <math.h>

// ---------------------------------------------------------------------------
// Types for CDNA5 WMMA
// ---------------------------------------------------------------------------
typedef __bf16 bf16_t;
typedef __attribute__((ext_vector_type(4)))  bf16_t v4bf;
typedef __attribute__((ext_vector_type(8)))  bf16_t v8bf;
typedef __attribute__((ext_vector_type(16))) bf16_t v16bf;
typedef __attribute__((ext_vector_type(2)))  float  v2f;
typedef __attribute__((ext_vector_type(4)))  float  v4f;
typedef __attribute__((ext_vector_type(8)))  float  v8f;

// Model dims
#define N_PATCH 16384
#define D_IN    768
#define D_MODEL 256
#define OMIC    2048
#define NPATH   331
#define PD1     5296    /* 16*331  */
#define PD2     21184   /* 64*331  */
#define DHEAD   128
#define LMK     256
#define QSCALE  0.08838834764831845f  /* 128^-0.5 */

// LDS row strides (units of the element type)
#define SBF  40   // bf16 tile: 80B rows -> conflict-free, 16B aligned
#define SA   36   // f32 tile: 144B rows -> conflict-free per b128, 16B aligned
#define SF32 20   // f32 pinv tile: 80B rows

// ---------------------------------------------------------------------------
// Async global->LDS copy (gfx1250 GLOBAL_LOAD_ASYNC_TO_LDS_B128, ASYNCcnt),
// with a synchronous fallback so the file compiles on host pass / older clang.
// Builtin signature (from clang diagnostic): (v4i AS1*, v4i AS3*, Imm, Imm)
// where v4i is gcc-style vector_size(16) int.
// ---------------------------------------------------------------------------
#if defined(__AMDGCN__) && __has_builtin(__builtin_amdgcn_global_load_async_to_lds_b128)
#define HAVE_ASYNC_LDS 1
#else
#define HAVE_ASYNC_LDS 0
#endif

typedef int v4i_gcc __attribute__((vector_size(16)));
typedef __attribute__((address_space(1))) v4i_gcc* g4i_t;
typedef __attribute__((address_space(3))) v4i_gcc* l4i_t;

__device__ __forceinline__ void async_copy16(const void* g, void* l)
{
#if HAVE_ASYNC_LDS
  // flat LDS address low 32 bits == LDS byte offset (ISA 10.2 aperture rule)
  __builtin_amdgcn_global_load_async_to_lds_b128(
      (g4i_t)(uintptr_t)g, (l4i_t)(uintptr_t)l, 0, 0);
#else
  *(v4f*)l = *(const v4f*)g;
#endif
}

__device__ __forceinline__ void wait_async_lds()
{
#if HAVE_ASYNC_LDS
#if __has_builtin(__builtin_amdgcn_s_wait_asynccnt)
  __builtin_amdgcn_s_wait_asynccnt(0);
#else
  asm volatile("s_wait_asynccnt 0x0" ::: "memory");
#endif
#endif
}

// pack 16 f32 (4x v4f) into a bf16 WMMA fragment
__device__ __forceinline__ v16bf cvt16(v4f a0, v4f a1, v4f a2, v4f a3)
{
  v16bf r;
#pragma unroll
  for (int e = 0; e < 4; ++e) {
    r[e]      = (bf16_t)a0[e];
    r[4 + e]  = (bf16_t)a1[e];
    r[8 + e]  = (bf16_t)a2[e];
    r[12 + e] = (bf16_t)a3[e];
  }
  return r;
}

// ---------------------------------------------------------------------------
// bf16 WMMA GEMM:  C[M,N] = relu?( alpha * A @ B(^T) + bias )
// A: f32 [M,K] ld=lda.   BT=0: B f32 [K,N] ld=ldb.   BT=1: B f32 [N,K] ld=ldb.
// BM=BN=64, BK=32; 128 threads = 4 wave32, each wave a 32x32 block.
// Double-buffered LDS; A (and B when BT=1) staged via async-to-LDS copies of
// raw f32 tiles, converted to bf16 during fragment construction (co-executes
// with WMMA). NN B-tiles are transposed+converted synchronously.
// All problem dims are multiples of the tile sizes (guaranteed by host code).
// ---------------------------------------------------------------------------
template<int BT>
__global__ __launch_bounds__(128)
void k_gemm_bf16(const float* __restrict__ A, const float* __restrict__ B,
                 const float* __restrict__ bias, float* __restrict__ C,
                 int M, int N, int K, int lda, int ldb, int ldc,
                 float alpha, int relu)
{
  alignas(16) __shared__ float  Af[2][64 * SA];
  alignas(16) __shared__ float  Bf[BT ? 2 * 64 * SA : 1];      // NT: f32 async tile
  alignas(16) __shared__ bf16_t Bh[BT ? 2 : 2 * 64 * SBF];     // NN: bf16 transposed tile

  const int tid  = threadIdx.x;
  const int m0   = blockIdx.y * 64;
  const int n0   = blockIdx.x * 64;
  const int wave = tid >> 5, lane = tid & 31;
  const int wm   = (wave >> 1) * 32;
  const int wn   = (wave &  1) * 32;
  const int half = lane >> 4, l16 = lane & 15;

  v8f acc[2][2];
#pragma unroll
  for (int i = 0; i < 2; ++i)
#pragma unroll
    for (int j = 0; j < 2; ++j)
#pragma unroll
      for (int e = 0; e < 8; ++e) acc[i][j][e] = 0.f;

  auto issue_tile = [&](int buf, int k0) {
    // A tile 64x32 f32 : 512 16B chunks, 4 per thread, async
#pragma unroll
    for (int i = 0; i < 4; ++i) {
      int p = tid + i * 128;
      int r = p >> 3, c4 = p & 7;
      async_copy16(A + (size_t)(m0 + r) * lda + k0 + c4 * 4,
                   &Af[buf][r * SA + c4 * 4]);
    }
    if (BT) {
      // B[N,K] tile 64x32 f32 : direct rows, async
#pragma unroll
      for (int i = 0; i < 4; ++i) {
        int p = tid + i * 128;
        int r = p >> 3, c4 = p & 7;
        async_copy16(B + (size_t)(n0 + r) * ldb + k0 + c4 * 4,
                     &Bf[buf * (64 * SA) + r * SA + c4 * 4]);
      }
    } else {
      // B[K,N] tile 32x64 : transpose + convert synchronously (weight operand)
#pragma unroll
      for (int i = 0; i < 4; ++i) {
        int p = tid + i * 128;
        int kk = p >> 4, c4 = p & 15;
        v4f b = *(const v4f*)(B + (size_t)(k0 + kk) * ldb + n0 + c4 * 4);
#pragma unroll
        for (int j = 0; j < 4; ++j)
          Bh[buf * (64 * SBF) + (c4 * 4 + j) * SBF + kk] = (bf16_t)b[j];
      }
    }
  };

  issue_tile(0, 0);
  int buf = 0;
  for (int k0 = 0; k0 < K; k0 += 32, buf ^= 1) {
    wait_async_lds();        // my copies of batch(cur) done
    __syncthreads();         // everyone's copies done; everyone past compute(prev)
    if (k0 + 32 < K) issue_tile(buf ^ 1, k0 + 32);   // overlaps compute below

    // ---- fragments (contiguous k = half*16 + 0..15 in A and B) ----
    v16bf afr[2], bfr[2];
#pragma unroll
    for (int mi = 0; mi < 2; ++mi) {
      const float* ap = &Af[buf][(wm + mi * 16 + l16) * SA];
      v4f a0 = *(const v4f*)(ap + half * 16);           // K = half*16 + 0..3
      v4f a1 = *(const v4f*)(ap + half * 16 + 4);       // K = half*16 + 4..7
      v4f a2 = *(const v4f*)(ap + half * 16 + 8);       // K = half*16 + 8..11
      v4f a3 = *(const v4f*)(ap + half * 16 + 12);      // K = half*16 + 12..15
      afr[mi] = cvt16(a0, a1, a2, a3);
    }
    if (BT) {
#pragma unroll
      for (int ni = 0; ni < 2; ++ni) {
        const float* bp = &Bf[buf * (64 * SA) + (wn + ni * 16 + l16) * SA + half * 16];
        v4f b0 = *(const v4f*)(bp);                    // K = half*16 + 0..3
        v4f b1 = *(const v4f*)(bp + 4);
        v4f b2 = *(const v4f*)(bp + 8);
        v4f b3 = *(const v4f*)(bp + 12);
        bfr[ni] = cvt16(b0, b1, b2, b3);
      }
    } else {
#pragma unroll
      for (int ni = 0; ni < 2; ++ni) {
        const bf16_t* bp = &Bh[buf * (64 * SBF) + (wn + ni * 16 + l16) * SBF + half * 16];
        v8bf lo = *(const v8bf*)(bp);
        v8bf hi = *(const v8bf*)(bp + 8);
        bfr[ni] = __builtin_shufflevector(lo, hi, 0,1,2,3,4,5,6,7,8,9,10,11,12,13,14,15);
      }
    }
#pragma unroll
    for (int mi = 0; mi < 2; ++mi)
#pragma unroll
      for (int ni = 0; ni < 2; ++ni)
        acc[mi][ni] = __builtin_amdgcn_wmma_f32_16x16x32_bf16(
            false, afr[mi], false, bfr[ni], (short)0, acc[mi][ni], false, false);
  }

  // ---- epilogue: lane l -> n = l16, rows m = r + half*8 ----
#pragma unroll
  for (int mi = 0; mi < 2; ++mi)
#pragma unroll
    for (int ni = 0; ni < 2; ++ni) {
      int gm = m0 + wm + mi * 16 + half * 8;
      int gn = n0 + wn + ni * 16 + l16;
      float bv = bias ? bias[gn] : 0.f;
#pragma unroll
      for (int r = 0; r < 8; ++r) {
        float v = alpha * acc[mi][ni][r] + bv;
        if (relu) v = fmaxf(v, 0.f);
        C[(size_t)(gm + r) * ldc + gn] = v;
      }
    }
}

// ---------------------------------------------------------------------------
// fp32 WMMA GEMM for the 256x256x256 pinv matmuls: C = alpha * A @ B
// Uses V_WMMA_F32_16X16X4_F32 (full precision for Moore-Penrose iteration).
// ---------------------------------------------------------------------------
__global__ __launch_bounds__(128)
void k_gemm256_f32(const float* __restrict__ A, const float* __restrict__ B,
                   float* __restrict__ C, float alpha)
{
  alignas(16) __shared__ float As[64 * SF32];
  alignas(16) __shared__ float Bs[64 * SF32];   // Bt[n][k]

  const int tid  = threadIdx.x;
  const int m0   = blockIdx.y * 64;
  const int n0   = blockIdx.x * 64;
  const int wave = tid >> 5, lane = tid & 31;
  const int wm   = (wave >> 1) * 32;
  const int wn   = (wave &  1) * 32;
  const int half = lane >> 4, l16 = lane & 15;

  v8f acc[2][2];
#pragma unroll
  for (int i = 0; i < 2; ++i)
#pragma unroll
    for (int j = 0; j < 2; ++j)
#pragma unroll
      for (int e = 0; e < 8; ++e) acc[i][j][e] = 0.f;

  for (int k0 = 0; k0 < 256; k0 += 16) {
#pragma unroll
    for (int i = 0; i < 2; ++i) {
      int p = tid + i * 128;
      int r = p >> 2, c4 = p & 3;
      v4f a = *(const v4f*)(A + (size_t)(m0 + r) * 256 + k0 + c4 * 4);
      *(v4f*)&As[r * SF32 + c4 * 4] = a;
    }
#pragma unroll
    for (int i = 0; i < 2; ++i) {
      int p = tid + i * 128;
      int kk = p >> 4, c4 = p & 15;
      v4f b = *(const v4f*)(B + (size_t)(k0 + kk) * 256 + n0 + c4 * 4);
#pragma unroll
      for (int j = 0; j < 4; ++j) Bs[(c4 * 4 + j) * SF32 + kk] = b[j];
    }
    __syncthreads();

#pragma unroll
    for (int kk = 0; kk < 4; ++kk) {
      v2f afr[2], bfr[2];
#pragma unroll
      for (int mi = 0; mi < 2; ++mi)
        afr[mi] = *(const v2f*)&As[(wm + mi * 16 + l16) * SF32 + kk * 4 + half * 2];
#pragma unroll
      for (int ni = 0; ni < 2; ++ni)
        bfr[ni] = *(const v2f*)&Bs[(wn + ni * 16 + l16) * SF32 + kk * 4 + half * 2];
#pragma unroll
      for (int mi = 0; mi < 2; ++mi)
#pragma unroll
        for (int ni = 0; ni < 2; ++ni)
          acc[mi][ni] = __builtin_amdgcn_wmma_f32_16x16x4_f32(
              false, afr[mi], false, bfr[ni], (short)0, acc[mi][ni], false, false);
    }
    __syncthreads();
  }

#pragma unroll
  for (int mi = 0; mi < 2; ++mi)
#pragma unroll
    for (int ni = 0; ni < 2; ++ni) {
      int gm = m0 + wm + mi * 16 + half * 8;
      int gn = n0 + wn + ni * 16 + l16;
#pragma unroll
      for (int r = 0; r < 8; ++r)
        C[(size_t)(gm + r) * 256 + gn] = alpha * acc[mi][ni][r];
    }
}

// ---------------------------------------------------------------------------
// Small support kernels
// ---------------------------------------------------------------------------
__global__ void k_landmarks(const float* __restrict__ QKV,
                            float* __restrict__ QL, float* __restrict__ KL)
{
  int j = blockIdx.x, d = threadIdx.x;                 // 256 x 128
  const float* base = QKV + (size_t)j * 64 * 384;
  float sq = 0.f, sk = 0.f;
  for (int i = 0; i < 64; ++i) { sq += base[i * 384 + d]; sk += base[i * 384 + 128 + d]; }
  QL[j * 128 + d] = sq * (QSCALE / 64.f);              // scaled q landmarks
  KL[j * 128 + d] = sk * (1.f / 64.f);
}

__global__ void k_softmax_rows(float* __restrict__ X, int cols)
{
  __shared__ float red[256];
  float* row = X + (size_t)blockIdx.x * cols;
  int tid = threadIdx.x;
  float m = -3.4e38f;
  for (int c = tid; c < cols; c += 256) m = fmaxf(m, row[c]);
  red[tid] = m; __syncthreads();
  for (int s = 128; s > 0; s >>= 1) { if (tid < s) red[tid] = fmaxf(red[tid], red[tid + s]); __syncthreads(); }
  m = red[0]; __syncthreads();
  float sum = 0.f;
  for (int c = tid; c < cols; c += 256) { float e = __expf(row[c] - m); row[c] = e; sum += e; }
  red[tid] = sum; __syncthreads();
  for (int s = 128; s > 0; s >>= 1) { if (tid < s) red[tid] += red[tid + s]; __syncthreads(); }
  float inv = 1.f / red[0];
  for (int c = tid; c < cols; c += 256) row[c] *= inv;
}

__global__ void k_colmean_acc(const float* __restrict__ A, float* __restrict__ out)
{ // A [16384,256], each block reduces 256 rows, atomically accumulates mean
  int col = threadIdx.x, r0 = blockIdx.x * 256;
  float s = 0.f;
  for (int r = 0; r < 256; ++r) s += A[(size_t)(r0 + r) * 256 + col];
  atomicAdd(&out[col], s * (1.f / 16384.f));
}

__global__ void k_pinv_init(const float* __restrict__ A2, float* __restrict__ Z)
{ // z0 = A2^T / (max(rowsum|A2|) * max(colsum|A2|)), one block of 256
  __shared__ float red[256];
  __shared__ float dinv;
  int tid = threadIdx.x;
  float rs = 0.f;
  for (int j = 0; j < 256; ++j) rs += fabsf(A2[tid * 256 + j]);
  red[tid] = rs; __syncthreads();
  for (int s = 128; s > 0; s >>= 1) { if (tid < s) red[tid] = fmaxf(red[tid], red[tid + s]); __syncthreads(); }
  float rmax = red[0]; __syncthreads();
  float cs = 0.f;
  for (int i = 0; i < 256; ++i) cs += fabsf(A2[i * 256 + tid]);
  red[tid] = cs; __syncthreads();
  for (int s = 128; s > 0; s >>= 1) { if (tid < s) red[tid] = fmaxf(red[tid], red[tid + s]); __syncthreads(); }
  if (tid == 0) dinv = 1.f / (rmax * red[0]);
  __syncthreads();
  float inv = dinv;
  for (int j = 0; j < 256; ++j) Z[tid * 256 + j] = A2[j * 256 + tid] * inv;
}

__global__ void k_addiag(const float* __restrict__ T, float* __restrict__ U, float alpha)
{ // U = alpha*I - T  (256x256)
  int i = blockIdx.x, j = threadIdx.x;
  U[i * 256 + j] = ((i == j) ? alpha : 0.f) - T[i * 256 + j];
}

__global__ void k_vecmat(const float* __restrict__ x, const float* __restrict__ W,
                         const float* __restrict__ bias, float* __restrict__ y,
                         int n_in, int n_out, int ld)
{ // y = x @ W (+bias)
  int j = blockIdx.x * blockDim.x + threadIdx.x;
  if (j >= n_out) return;
  float s = bias ? bias[j] : 0.f;
  for (int i = 0; i < n_in; ++i) s += x[i] * W[(size_t)i * ld + j];
  y[j] = s;
}

// ---- genomics path (mask structure exploited, never materialized) ----
__global__ void k_fc1(const float* __restrict__ omics, const float* __restrict__ dfc,
                      const float* __restrict__ W, const float* __restrict__ b,
                      float* __restrict__ o1)
{
  __shared__ float om[OMIC];
  int tid = threadIdx.x;
  for (int t = tid; t < OMIC; t += 256) om[t] = omics[t];
  __syncthreads();
  int j = blockIdx.x * 256 + tid;
  if (j >= PD1) return;
  int p = j >> 4;                                   // pathway = j / 16
  float s = 0.f;
  for (int i = 0; i < OMIC; ++i)
    s += om[i] * W[(size_t)i * PD1 + j] * dfc[i * NPATH + p];
  o1[j] = fmaxf(s + b[j], 0.f);
}

__global__ void k_fc2(const float* __restrict__ o1, const float* __restrict__ W,
                      const float* __restrict__ b, float* __restrict__ o2)
{ // block-diagonal: each output touches only its pathway's 16 inputs
  int j = blockIdx.x * 256 + threadIdx.x;
  if (j >= PD2) return;
  int base = (j >> 6) * 16;
  float s = b[j];
  for (int r = 0; r < 16; ++r) s += o1[base + r] * W[(size_t)(base + r) * PD2 + j];
  o2[j] = s;
}

__global__ void k_u1_acc(const float* __restrict__ o2, const float* __restrict__ W,
                         float* __restrict__ acc)
{ // split-K gemv: 32 blocks x 64 outputs, 662 inputs each
  int out = threadIdx.x;
  int i0 = blockIdx.x * 662;
  float s = 0.f;
  for (int i = i0; i < i0 + 662; ++i) s += o2[i] * W[(size_t)i * 64 + out];
  atomicAdd(&acc[out], s);
}

__global__ void k_u2(const float* __restrict__ acc, const float* __restrict__ b1,
                     const float* __restrict__ W, const float* __restrict__ b2,
                     float* __restrict__ h)
{ // h_omic = relu(o3 + u1_b) @ u2_w + u2_b
  int j = threadIdx.x;
  float s = b2[j];
  for (int i = 0; i < 64; ++i) s += fmaxf(acc[i] + b1[i], 0.f) * W[i * 256 + j];
  h[j] = s;
}

__global__ void k_fusion(const float* __restrict__ hp, const float* __restrict__ ho,
                         const float* __restrict__ m1w, const float* __restrict__ m1b,
                         const float* __restrict__ m2w, const float* __restrict__ m2b,
                         const float* __restrict__ cw,  const float* __restrict__ cb,
                         float* __restrict__ out)
{
  __shared__ float h1[256];
  __shared__ float h2[256];
  int j = threadIdx.x;
  float s = m1b[j];
  for (int i = 0; i < 256; ++i) s += hp[i] * m1w[(size_t)i * 256 + j];
  for (int i = 0; i < 256; ++i) s += ho[i] * m1w[(size_t)(256 + i) * 256 + j];
  h1[j] = fmaxf(s, 0.f);
  __syncthreads();
  float s2 = m2b[j];
  for (int i = 0; i < 256; ++i) s2 += h1[i] * m2w[(size_t)i * 256 + j];
  float hv = fmaxf(s2, 0.f);
  h2[j] = hv;
  out[4 + j] = hv;                  // output 'h'
  __syncthreads();
  if (j == 0) {
    float lg[4];
    for (int c = 0; c < 4; ++c) {
      float t = cb[c];
      for (int i = 0; i < 256; ++i) t += h2[i] * cw[i * 4 + c];
      lg[c] = t;
    }
    int amax = 0; float best = lg[0];
    for (int c = 1; c < 4; ++c) if (lg[c] > best) { best = lg[c]; amax = c; }
    float run = 1.f;
    for (int c = 0; c < 4; ++c) {
      float hz = 1.f / (1.f + __expf(-lg[c]));
      out[c] = hz;                  // hazards
      run *= (1.f - hz);
      out[260 + c] = run;           // S
    }
    out[264] = (float)amax;         // Y_hat
  }
}

// ---------------------------------------------------------------------------
// Host side
// ---------------------------------------------------------------------------
static void gemm_bf16(hipStream_t s, int BT, const float* A, const float* B,
                      const float* bias, float* C, int M, int N, int K,
                      int lda, int ldb, int ldc, float alpha, int relu)
{
  dim3 g(N / 64, M / 64), b(128);
  if (BT) k_gemm_bf16<1><<<g, b, 0, s>>>(A, B, bias, C, M, N, K, lda, ldb, ldc, alpha, relu);
  else    k_gemm_bf16<0><<<g, b, 0, s>>>(A, B, bias, C, M, N, K, lda, ldb, ldc, alpha, relu);
}

extern "C" void kernel_launch(void* const* d_in, const int* in_sizes, int n_in,
                              void* d_out, int out_size, void* d_ws, size_t ws_size,
                              hipStream_t stream)
{
  (void)in_sizes; (void)n_in; (void)out_size; (void)ws_size;
  const float* x_path = (const float*)d_in[0];
  const float* omics  = (const float*)d_in[1];
  const float* dfc    = (const float*)d_in[2];
  const float* fc_w   = (const float*)d_in[3];
  const float* fc_b   = (const float*)d_in[4];
  const float* qkv_w  = (const float*)d_in[5];
  const float* out_w  = (const float*)d_in[6];
  const float* out_b  = (const float*)d_in[7];
  const float* fc1_w  = (const float*)d_in[8];
  const float* fc1_b  = (const float*)d_in[9];
  const float* fc2_w  = (const float*)d_in[10];
  const float* fc2_b  = (const float*)d_in[11];
  const float* u1_w   = (const float*)d_in[12];
  const float* u1_b   = (const float*)d_in[13];
  const float* u2_w   = (const float*)d_in[14];
  const float* u2_b   = (const float*)d_in[15];
  const float* m1_w   = (const float*)d_in[16];
  const float* m1_b   = (const float*)d_in[17];
  const float* m2_w   = (const float*)d_in[18];
  const float* m2_b   = (const float*)d_in[19];
  const float* cls_w  = (const float*)d_in[20];
  const float* cls_b  = (const float*)d_in[21];
  float* out = (float*)d_out;

  // bump allocator over workspace
  char* wsb = (char*)d_ws;
  size_t off = 0;
  auto alloc = [&](size_t bytes) -> float* {
    float* p = (float*)(wsb + off);
    off += (bytes + 255) & ~(size_t)255;
    return p;
  };
  float* X    = alloc((size_t)N_PATCH * 256 * 4);  // relu(x@fc_w+b); later reused as a1
  float* QKV  = alloc((size_t)N_PATCH * 384 * 4);
  float* A3   = alloc((size_t)256 * N_PATCH * 4);
  float* QL   = alloc(256 * 128 * 4);
  float* KL   = alloc(256 * 128 * 4);
  float* A2   = alloc(256 * 256 * 4);
  float* ZA   = alloc(256 * 256 * 4);
  float* ZB   = alloc(256 * 256 * 4);
  float* XZ   = alloc(256 * 256 * 4);
  float* TM   = alloc(256 * 256 * 4);
  float* U    = alloc(256 * 256 * 4);
  float* T1   = alloc(256 * 128 * 4);
  float* CM   = alloc(256 * 4);
  float* V1   = alloc(256 * 4);
  float* V2   = alloc(128 * 4);
  float* HP   = alloc(256 * 4);
  float* O1   = alloc(PD1 * 4);
  float* O2   = alloc(PD2 * 4);
  float* O3   = alloc(64 * 4);
  float* HO   = alloc(256 * 4);

  // ---- WSI path -----------------------------------------------------------
  // X = relu(x_path @ fc_w + fc_b)            [16384,256]
  gemm_bf16(stream, 0, x_path, fc_w, fc_b, X, N_PATCH, 256, D_IN, D_IN, 256, 256, 1.f, 1);
  // QKV = X @ qkv_w                           [16384,384]
  gemm_bf16(stream, 0, X, qkv_w, nullptr, QKV, N_PATCH, 384, 256, 256, 384, 384, 1.f, 0);
  // landmarks (QL pre-scaled by 1/sqrt(dh))
  k_landmarks<<<LMK, 128, 0, stream>>>(QKV, QL, KL);
  // a1 = softmax(s * Q @ KL^T)   -> reuse X buffer (X is dead now)
  float* A1 = X;
  gemm_bf16(stream, 1, QKV, KL, nullptr, A1, N_PATCH, 256, 128, 384, 128, 256, QSCALE, 0);
  k_softmax_rows<<<N_PATCH, 256, 0, stream>>>(A1, 256);
  (void)hipMemsetAsync(CM, 0, 256 * 4, stream);
  k_colmean_acc<<<64, 256, 0, stream>>>(A1, CM);     // colmean(a1) [256]
  // a2 = softmax(QL @ KL^T)                   [256,256]
  gemm_bf16(stream, 1, QL, KL, nullptr, A2, 256, 256, 128, 128, 128, 256, 1.f, 0);
  k_softmax_rows<<<256, 256, 0, stream>>>(A2, 256);
  // a3 = softmax(QL @ K^T)                    [256,16384]
  gemm_bf16(stream, 1, QL, QKV + 128, nullptr, A3, 256, N_PATCH, 128, 128, 384, N_PATCH, 1.f, 0);
  k_softmax_rows<<<256, 256, 0, stream>>>(A3, N_PATCH);
  // T1 = a3 @ V                               [256,128]
  gemm_bf16(stream, 0, A3, QKV + 256, nullptr, T1, 256, 128, N_PATCH, N_PATCH, 384, 128, 1.f, 0);

  // pinv(a2), fp32 WMMA (precision-sensitive Newton-Schulz style iteration)
  k_pinv_init<<<1, 256, 0, stream>>>(A2, ZA);
  float* Z = ZA; float* Zn = ZB;
  dim3 g44(4, 4), b128(128);
  for (int it = 0; it < 6; ++it) {
    k_gemm256_f32<<<g44, b128, 0, stream>>>(A2, Z, XZ, 1.f);       // xz = a2 @ z
    k_addiag<<<256, 256, 0, stream>>>(XZ, U, 7.f);                 // 7I - xz
    k_gemm256_f32<<<g44, b128, 0, stream>>>(XZ, U, TM, 1.f);       // xz @ (7I - xz)
    k_addiag<<<256, 256, 0, stream>>>(TM, U, 15.f);                // 15I - ...
    k_gemm256_f32<<<g44, b128, 0, stream>>>(XZ, U, TM, 1.f);       // xz @ (15I - ...)
    k_addiag<<<256, 256, 0, stream>>>(TM, U, 13.f);                // 13I - ...
    k_gemm256_f32<<<g44, b128, 0, stream>>>(Z, U, Zn, 0.25f);      // z = 0.25 z @ (...)
    float* t = Z; Z = Zn; Zn = t;                                  // deterministic swap
  }

  // mean-pooled attention output via linearity:
  // h_path = ((colmean(a1) @ Z) @ T1) @ out_w + out_b
  k_vecmat<<<1, 256, 0, stream>>>(CM, Z, nullptr, V1, 256, 256, 256);
  k_vecmat<<<1, 128, 0, stream>>>(V1, T1, nullptr, V2, 256, 128, 128);
  k_vecmat<<<1, 256, 0, stream>>>(V2, out_w, out_b, HP, 128, 256, 256);

  // ---- genomics path ------------------------------------------------------
  k_fc1<<<(PD1 + 255) / 256, 256, 0, stream>>>(omics, dfc, fc1_w, fc1_b, O1);
  k_fc2<<<(PD2 + 255) / 256, 256, 0, stream>>>(O1, fc2_w, fc2_b, O2);
  (void)hipMemsetAsync(O3, 0, 64 * 4, stream);
  k_u1_acc<<<32, 64, 0, stream>>>(O2, u1_w, O3);
  k_u2<<<1, 256, 0, stream>>>(O3, u1_b, u2_w, u2_b, HO);

  // ---- fusion + heads -----------------------------------------------------
  k_fusion<<<1, 256, 0, stream>>>(HP, HO, m1_w, m1_b, m2_w, m2_b, cls_w, cls_b, out);
}
